// SparseBlock_41240275976491
// MI455X (gfx1250) — matrix operand order repaired and checked by reading, not compiled, not run on hardware
//
#include <hip/hip_runtime.h>

typedef __bf16 bf16;
typedef bf16  v16bf __attribute__((ext_vector_type(16)));
typedef float v8f   __attribute__((ext_vector_type(8)));

#define B_    4
#define CIN   128
#define H_    256
#define W_    256
#define HW    65536          // H_*W_
#define COUT  256
#define NB    128            // active blocks
#define BS_   32
#define PPB   1024           // positions per block (32*32)
#define NTOT  (NB*PPB)       // 131072
#define EPS_  1e-5f

// ---- workspace layout (bytes) ----
constexpr size_t OFF_WD = 0;                                   // 256*128 bf16
constexpr size_t OFF_W1 = OFF_WD + (size_t)COUT*CIN*2;         // 65536
constexpr size_t OFF_W2 = OFF_W1 + (size_t)COUT*CIN*2;         // 131072, [9][256][256] bf16
constexpr size_t OFF_BN = OFF_W2 + (size_t)9*COUT*COUT*2;      // 1310720, 8*256 f32
constexpr size_t OFF_H1 = OFF_BN + (size_t)8*COUT*4;           // 1318912, [128][1024][256] bf16
constexpr size_t OFF_H2 = OFF_H1 + (size_t)NB*COUT*PPB*2;      // +64MB, [128][256][1024] bf16

// ---------------------------------------------------------------------------
// Weight prep: f32 -> bf16, w2 re-laid out as [tap][o][c]
// ---------------------------------------------------------------------------
__global__ __launch_bounds__(256) void k_prep_weights(
    const float* __restrict__ w1, const float* __restrict__ w2,
    const float* __restrict__ wd,
    bf16* __restrict__ wdb, bf16* __restrict__ w1b, bf16* __restrict__ w2b)
{
    const int NWD = COUT * CIN;   // 32768
    int idx = blockIdx.x * 256 + threadIdx.x;
    if (idx < NWD) {
        wdb[idx] = (bf16)wd[idx];
    } else if (idx < 2 * NWD) {
        w1b[idx - NWD] = (bf16)w1[idx - NWD];
    } else {
        int j = idx - 2 * NWD;                       // index into w2 [o][c][3][3]
        if (j < 9 * COUT * COUT) {
            int o    = j / (COUT * 9);
            int rem  = j - o * (COUT * 9);
            int c    = rem / 9;
            int tap  = rem - c * 9;
            w2b[tap * (COUT * COUT) + o * COUT + c] = (bf16)w2[j];
        }
    }
}

__global__ void k_zero_bn(float* __restrict__ bn)
{
    bn[blockIdx.x * 256 + threadIdx.x] = 0.f;   // grid=4 -> sums1,sq1,sums2,sq2
}

// ---------------------------------------------------------------------------
// Dense base path: out[b,o,p] = sum_k wd[o,k]*x[b,k,p] + bd[o]
// Tile: M=128 (8 waves x 16), N=64, K=128.  bf16 WMMA 16x16x32, f32 acc.
// ---------------------------------------------------------------------------
__global__ __launch_bounds__(256) void k_base_gemm(
    const float* __restrict__ x, const bf16* __restrict__ wdb,
    const float* __restrict__ bd, float* __restrict__ out)
{
    __shared__ __align__(16) bf16 Wl[128][136];   // [m][k]
    __shared__ __align__(16) bf16 Xl[64][136];    // [n][k] (transposed)

    const int tid = threadIdx.x;
    const int m0  = blockIdx.y * 128;
    const int pg  = blockIdx.x * 64;             // global column over B*HW
    const int b   = pg >> 16;                    // / HW
    const int p   = pg & (HW - 1);

    for (int i = 0; i < 64; ++i) {               // 128x128 weights
        int idx = tid + i * 256;
        int m = idx >> 7, k = idx & 127;
        Wl[m][k] = wdb[(m0 + m) * CIN + k];
    }
    for (int i = 0; i < 32; ++i) {               // 64x128 activations
        int idx = tid + i * 256;
        int n = idx & 63, k = idx >> 6;
        Xl[n][k] = (bf16)x[b * (CIN * HW) + k * HW + p + n];
    }
    __syncthreads();

    const int wave = tid >> 5, lane = tid & 31;
    const int mrow = lane & 15;
    const int kbA  = (lane < 16) ? 0 : 16;       // contiguous k order (k = 16h + s)
    const int kbB  = (lane < 16) ? 0 : 16;       // same contiguous k order as A

    v8f acc[4];
    for (int j = 0; j < 4; ++j)
        for (int r = 0; r < 8; ++r) acc[j][r] = 0.f;

#pragma unroll
    for (int kb = 0; kb < 128; kb += 32) {
        v16bf a;
        const bf16* pa = &Wl[wave * 16 + mrow][kb + kbA];
#pragma unroll
        for (int i = 0; i < 16; ++i) a[i] = pa[i];
#pragma unroll
        for (int j = 0; j < 4; ++j) {
            v16bf bb;
            const bf16* pb = &Xl[j * 16 + mrow][kb + kbB];
#pragma unroll
            for (int i = 0; i < 16; ++i) bb[i] = pb[i];
            acc[j] = __builtin_amdgcn_wmma_f32_16x16x32_bf16(
                false, a, false, bb, (short)0, acc[j], false, false);
        }
    }

    const int mhi = (lane < 16) ? 0 : 8;
#pragma unroll
    for (int j = 0; j < 4; ++j) {
#pragma unroll
        for (int r = 0; r < 8; ++r) {
            int o  = m0 + wave * 16 + r + mhi;
            int pc = p + j * 16 + (lane & 15);
            out[b * (COUT * HW) + o * HW + pc] = acc[j][r] + bd[o];
        }
    }
}

// ---------------------------------------------------------------------------
// Gather + 1x1 conv + bias -> h1 bf16 [nb][p][c] (channel-contiguous),
// plus BN1 partial sums from the f32 accumulators.
// ---------------------------------------------------------------------------
__global__ __launch_bounds__(256) void k_conv1(
    const float* __restrict__ x, const bf16* __restrict__ w1b,
    const float* __restrict__ b1, const int* __restrict__ bidx,
    bf16* __restrict__ h1, float* __restrict__ gsum, float* __restrict__ gsq)
{
    __shared__ __align__(16) bf16 Wl[128][136];
    __shared__ __align__(16) bf16 Xl[64][136];
    __shared__ float lsum[128], lsq[128];

    const int tid = threadIdx.x;
    const int m0  = blockIdx.y * 128;
    const int nb  = blockIdx.x >> 4;
    const int off = (blockIdx.x & 15) * 64;
    const int bi = bidx[nb * 3 + 0], by = bidx[nb * 3 + 1], bx = bidx[nb * 3 + 2];
    if (tid < 128) { lsum[tid] = 0.f; lsq[tid] = 0.f; }

    for (int i = 0; i < 64; ++i) {
        int idx = tid + i * 256;
        int m = idx >> 7, k = idx & 127;
        Wl[m][k] = w1b[(m0 + m) * CIN + k];
    }
    for (int i = 0; i < 32; ++i) {
        int idx = tid + i * 256;
        int n = idx & 63, k = idx >> 6;
        int pib = off + n;
        int yy = pib >> 5, xx = pib & 31;
        Xl[n][k] = (bf16)x[bi * (CIN * HW) + k * HW + (by * BS_ + yy) * W_ + bx * BS_ + xx];
    }
    __syncthreads();

    const int wave = tid >> 5, lane = tid & 31;
    const int mrow = lane & 15;
    const int kbA  = (lane < 16) ? 0 : 16;
    const int kbB  = (lane < 16) ? 0 : 16;

    v8f acc[4];
    for (int j = 0; j < 4; ++j)
        for (int r = 0; r < 8; ++r) acc[j][r] = 0.f;

#pragma unroll
    for (int kb = 0; kb < 128; kb += 32) {
        v16bf a;
        const bf16* pa = &Wl[wave * 16 + mrow][kb + kbA];
#pragma unroll
        for (int i = 0; i < 16; ++i) a[i] = pa[i];
#pragma unroll
        for (int j = 0; j < 4; ++j) {
            v16bf bb;
            const bf16* pb = &Xl[j * 16 + mrow][kb + kbB];
#pragma unroll
            for (int i = 0; i < 16; ++i) bb[i] = pb[i];
            acc[j] = __builtin_amdgcn_wmma_f32_16x16x32_bf16(
                false, a, false, bb, (short)0, acc[j], false, false);
        }
    }

    const int mhi = (lane < 16) ? 0 : 8;
    const int o0  = m0 + wave * 16 + mhi;       // multiple of 8 -> 16B-aligned stores
#pragma unroll
    for (int j = 0; j < 4; ++j) {
        int pc = off + j * 16 + (lane & 15);
        __attribute__((aligned(16))) bf16 pack[8];
#pragma unroll
        for (int r = 0; r < 8; ++r) {
            float v = acc[j][r] + b1[o0 + r];
            pack[r] = (bf16)v;
            atomicAdd(&lsum[o0 - m0 + r], v);
            atomicAdd(&lsq[o0 - m0 + r], v * v);
        }
        // h1 layout: [nb][p][c]; 8 contiguous channels -> one b128 store
        *(uint4*)(h1 + (size_t)nb * (COUT * PPB) + (size_t)pc * COUT + o0) =
            *(const uint4*)pack;
    }
    __syncthreads();
    if (tid < 128) {
        atomicAdd(&gsum[m0 + tid], lsum[tid]);
        atomicAdd(&gsq[m0 + tid], lsq[tid]);
    }
}

// ---------------------------------------------------------------------------
// Fold BN stats into per-channel scale/shift
// ---------------------------------------------------------------------------
__global__ void k_bn_finalize(
    const float* __restrict__ sum, const float* __restrict__ sq,
    const float* __restrict__ g, const float* __restrict__ be,
    float* __restrict__ scale, float* __restrict__ shift, float invN)
{
    int c = threadIdx.x;                       // 256 threads
    float mu  = sum[c] * invN;
    float var = sq[c] * invN - mu * mu;
    float rs  = rsqrtf(var + EPS_);
    float sc  = g[c] * rs;
    scale[c] = sc;
    shift[c] = be[c] - mu * sc;
}

// ---------------------------------------------------------------------------
// Apply relu(bn1(.)) to h1 in place (h1 layout [nb][p][c] => c == threadIdx.x)
// Makes conv2's LDS staging a raw copy -> async-LDS eligible.
// ---------------------------------------------------------------------------
__global__ __launch_bounds__(256) void k_bn1_apply(
    bf16* __restrict__ h1, const float* __restrict__ sc1,
    const float* __restrict__ sh1)
{
    const size_t base = (size_t)blockIdx.x * 1024;
    const float sc = sc1[threadIdx.x], sh = sh1[threadIdx.x];
#pragma unroll
    for (int i = 0; i < 4; ++i) {
        size_t e = base + threadIdx.x + i * 256;
        float v = sc * (float)h1[e] + sh;
        v = v > 0.f ? v : 0.f;
        h1[e] = (bf16)v;
    }
}

// ---------------------------------------------------------------------------
// 3x3 conv per block (SAME, zero halo) on preactivated h1 -> h2 + BN2 sums.
// Halo tile staged with GLOBAL_LOAD_ASYNC_TO_LDS_B128 (ASYNCcnt path).
// Tile: one block, 4 output rows (N=128), M=128, K=256 split in 2 LDS passes.
// ---------------------------------------------------------------------------
__global__ __launch_bounds__(256) void k_conv2(
    const bf16* __restrict__ h1, const bf16* __restrict__ w2b,
    const float* __restrict__ b2, bf16* __restrict__ h2,
    float* __restrict__ gsum, float* __restrict__ gsq)
{
    __shared__ __align__(16) bf16 Il[204][136];   // 6 halo rows x 34 cols, 128 ch
    __shared__ float lsum[128], lsq[128];

    const int tid = threadIdx.x;
    const int m0  = blockIdx.y * 128;
    const int nb  = blockIdx.x >> 3;
    const int y0  = (blockIdx.x & 7) * 4;
    if (tid < 128) { lsum[tid] = 0.f; lsq[tid] = 0.f; }

    const int wave = tid >> 5, lane = tid & 31;
    const int mrow = lane & 15;
    const int kbA  = (lane < 16) ? 0 : 16;
    const int kbB  = (lane < 16) ? 0 : 16;

    // 32-bit LDS byte offset of Il (generic LDS address: low 32 bits = LDS addr)
    const unsigned ldsbase = (unsigned)(size_t)(&Il[0][0]);
    char* Ib = (char*)&Il[0][0];

    v8f acc[8];
    for (int j = 0; j < 8; ++j)
        for (int r = 0; r < 8; ++r) acc[j][r] = 0.f;

    for (int kh = 0; kh < 2; ++kh) {
        __syncthreads();
        // --- stage halo tile: raw bf16 copy, 16B granules, direct to LDS ---
        // 204 halo positions x 128 channels x 2B = 3264 x 16B granules
        for (int it = 0; it < 13; ++it) {
            int idx = tid + it * 256;
            if (idx < 3264) {
                int hp = idx >> 4, kc = idx & 15;
                int hy = hp / 34, hx = hp - hy * 34;
                int iy = y0 + hy - 1, ix = hx - 1;
                unsigned lds = ldsbase + (unsigned)(hp * 272 + kc * 16);
                if (iy >= 0 && iy < BS_ && ix >= 0 && ix < BS_) {
                    const char* g = (const char*)h1 +
                        (((size_t)nb * (COUT * PPB) +
                          (size_t)(iy * BS_ + ix) * COUT + kh * 128) * 2 +
                         (size_t)kc * 16);
                    asm volatile("global_load_async_to_lds_b128 %0, %1, off"
                                 :: "v"(lds), "v"(g) : "memory");
                } else {
                    uint4 z; z.x = 0; z.y = 0; z.z = 0; z.w = 0;
                    *(uint4*)(Ib + hp * 272 + kc * 16) = z;   // zero halo cell
                }
            }
        }
        asm volatile("s_wait_asynccnt 0x0" ::: "memory");
        __syncthreads();

        for (int tap = 0; tap < 9; ++tap) {
            int dy = tap / 3, dx = tap - dy * 3;
            const bf16* wtap = w2b + tap * (COUT * COUT)
                             + (m0 + wave * 16 + mrow) * COUT + kh * 128;
            // prefetch next tap's weight row into caches (global_prefetch_b8)
            __builtin_prefetch(w2b + ((tap + 1) % 9) * (COUT * COUT)
                               + (m0 + wave * 16 + mrow) * COUT + kh * 128, 0, 3);
#pragma unroll
            for (int kb = 0; kb < 128; kb += 32) {
                v16bf a;
                const bf16* pa = wtap + kb + kbA;
#pragma unroll
                for (int i = 0; i < 16; ++i) a[i] = pa[i];
#pragma unroll
                for (int j = 0; j < 8; ++j) {
                    int n  = j * 16 + mrow;
                    int hp = ((n >> 5) + dy) * 34 + (n & 31) + dx;
                    v16bf bb;
                    const bf16* pb = &Il[hp][kb + kbB];
#pragma unroll
                    for (int i = 0; i < 16; ++i) bb[i] = pb[i];
                    acc[j] = __builtin_amdgcn_wmma_f32_16x16x32_bf16(
                        false, a, false, bb, (short)0, acc[j], false, false);
                }
            }
        }
    }

    const int mhi = (lane < 16) ? 0 : 8;
#pragma unroll
    for (int j = 0; j < 8; ++j) {
        int n  = j * 16 + mrow;
        int yy = n >> 5, xx = n & 31;
#pragma unroll
        for (int r = 0; r < 8; ++r) {
            int o = m0 + wave * 16 + r + mhi;
            float v = acc[j][r] + b2[o];
            h2[(size_t)nb * (COUT * PPB) + o * PPB + (y0 + yy) * BS_ + xx] = (bf16)v;
            atomicAdd(&lsum[o - m0], v);
            atomicAdd(&lsq[o - m0], v * v);
        }
    }
    __syncthreads();
    if (tid < 128) {
        atomicAdd(&gsum[m0 + tid], lsum[tid]);
        atomicAdd(&gsq[m0 + tid], lsq[tid]);
    }
}

// ---------------------------------------------------------------------------
// Scatter: relu(bn2(h2)) overwrites active blocks in the base output
// ---------------------------------------------------------------------------
__global__ __launch_bounds__(256) void k_scatter(
    const bf16* __restrict__ h2, const float* __restrict__ sc2,
    const float* __restrict__ sh2, const int* __restrict__ bidx,
    float* __restrict__ out)
{
    const int base = blockIdx.x * 1024;          // [nb][c][p], c uniform per wg
    const int nb = base >> 18;
    const int c  = (base >> 10) & 255;
    const int bi = bidx[nb * 3 + 0], by = bidx[nb * 3 + 1], bx = bidx[nb * 3 + 2];
    const float sc = sc2[c], sh = sh2[c];
#pragma unroll
    for (int i = 0; i < 4; ++i) {
        int p = threadIdx.x + i * 256;
        float v = sc * (float)h2[base + p] + sh;
        v = v > 0.f ? v : 0.f;
        int y = p >> 5, xx = p & 31;
        out[bi * (COUT * HW) + c * HW + (by * BS_ + y) * W_ + bx * BS_ + xx] = v;
    }
}

// ---------------------------------------------------------------------------
extern "C" void kernel_launch(void* const* d_in, const int* in_sizes, int n_in,
                              void* d_out, int out_size, void* d_ws, size_t ws_size,
                              hipStream_t stream)
{
    const float* x   = (const float*)d_in[0];
    const float* w1  = (const float*)d_in[1];
    const float* b1  = (const float*)d_in[2];
    const float* g1  = (const float*)d_in[3];
    const float* be1 = (const float*)d_in[4];
    const float* w2  = (const float*)d_in[5];
    const float* b2  = (const float*)d_in[6];
    const float* g2  = (const float*)d_in[7];
    const float* be2 = (const float*)d_in[8];
    const float* wd  = (const float*)d_in[9];
    const float* bd  = (const float*)d_in[10];
    const int*  bidx = (const int*)d_in[11];
    float* out = (float*)d_out;

    char* ws  = (char*)d_ws;
    bf16* wdb = (bf16*)(ws + OFF_WD);
    bf16* w1b = (bf16*)(ws + OFF_W1);
    bf16* w2b = (bf16*)(ws + OFF_W2);
    float* bn = (float*)(ws + OFF_BN);
    float *sum1 = bn,        *sq1 = bn + 256;
    float *sum2 = bn + 512,  *sq2 = bn + 768;
    float *sc1  = bn + 1024, *sh1 = bn + 1280;
    float *sc2  = bn + 1536, *sh2 = bn + 1792;
    bf16* h1 = (bf16*)(ws + OFF_H1);
    bf16* h2 = (bf16*)(ws + OFF_H2);

    const float invN = 1.0f / (float)NTOT;

    k_prep_weights<<<2560, 256, 0, stream>>>(w1, w2, wd, wdb, w1b, w2b);
    k_zero_bn<<<4, 256, 0, stream>>>(bn);
    k_base_gemm<<<dim3((B_ * HW) / 64, COUT / 128), 256, 0, stream>>>(x, wdb, bd, out);
    k_conv1<<<dim3(NTOT / 64, COUT / 128), 256, 0, stream>>>(x, w1b, b1, bidx, h1, sum1, sq1);
    k_bn_finalize<<<1, 256, 0, stream>>>(sum1, sq1, g1, be1, sc1, sh1, invN);
    k_bn1_apply<<<(NB * COUT * PPB) / 1024, 256, 0, stream>>>(h1, sc1, sh1);
    k_conv2<<<dim3(NB * 8, COUT / 128), 256, 0, stream>>>(h1, w2b, b2, h2, sum2, sq2);
    k_bn_finalize<<<1, 256, 0, stream>>>(sum2, sq2, g2, be2, sc2, sh2, invN);
    k_scatter<<<(NB * COUT * PPB) / 1024, 256, 0, stream>>>(h2, sc2, sh2, bidx, out);
}
